// SingleConstFilteredNormalized_30528627540479
// MI455X (gfx1250) — compile-verified, matrix operand order checked by reading, not run on hardware
//
#include <hip/hip_runtime.h>
#include <hip/hip_bf16.h>

typedef __attribute__((ext_vector_type(16))) _Float16 v16h;
typedef __attribute__((ext_vector_type(8)))  float    v8f;

#define ROW_D          8192
#define TPB            256
#define F4_PER_THREAD  8   // (8192 / 4) / 256

// One block per row. Phase 1: stream the row in with coalesced b128 loads,
// compress to a 32-bit mask per thread + popcount. Phase 2: reduce the 256
// partial counts with one V_WMMA_F32_16X16X32_F16 against an all-ones B
// (row-sum trick: with B=1, the grand total is independent of A packing).
// Phase 3: stream the output out with b128 stores.
__global__ __launch_bounds__(TPB)
void rownorm_mask_wmma_kernel(const float* __restrict__ x,
                              const float* __restrict__ f,
                              float* __restrict__ y)
{
    __shared__ __align__(64) _Float16 cnt[512];  // 256 partials + 256 zero pad
    __shared__ float bcast;

    const int tid = threadIdx.x;
    const int row = blockIdx.x;

    const float4* __restrict__ xr = (const float4*)(x + (size_t)row * ROW_D);
    float4*       __restrict__ yr = (float4*)(y + (size_t)row * ROW_D);

    // ---- Phase 1: coalesced streaming read, compress to mask bits ----
    float4 v[F4_PER_THREAD];
#pragma unroll
    for (int k = 0; k < F4_PER_THREAD; ++k)
        v[k] = xr[k * TPB + tid];           // lanes contiguous per instruction

    unsigned m = 0u;
#pragma unroll
    for (int k = 0; k < F4_PER_THREAD; ++k) {
        m |= (v[k].x != 0.0f ? 1u : 0u) << (4 * k + 0);
        m |= (v[k].y != 0.0f ? 1u : 0u) << (4 * k + 1);
        m |= (v[k].z != 0.0f ? 1u : 0u) << (4 * k + 2);
        m |= (v[k].w != 0.0f ? 1u : 0u) << (4 * k + 3);
    }
    const int pc = __popc(m);               // 0..32, exact in f16

    cnt[tid]       = (_Float16)pc;
    cnt[256 + tid] = (_Float16)0.0f;        // zero pad so A is a full 16x32
    __syncthreads();

    // ---- Phase 2: block reduction via WMMA (wave 0 only; EXEC all-1s) ----
    if (tid < 32) {
        v16h a = *(const v16h*)&cnt[tid * 16];   // any bijection into A is fine
        v16h b;
#pragma unroll
        for (int i = 0; i < 16; ++i) b[i] = (_Float16)1.0f;
        v8f c = {};
        // D[i][j] = sum_k A[i][k] * 1.0  (f32 accumulate, exact)
        c = __builtin_amdgcn_wmma_f32_16x16x32_f16(
                /*neg_a=*/false, a, /*neg_b=*/false, b,
                /*c_mod=*/(short)0, c, /*reuse_a=*/false, /*reuse_b=*/false);
        // Per lane: 8 VGPRs hold row-sums for M=0..7 (lanes 0-15) or M=8..15
        // (lanes 16-31). Fold VGPRs, then fold the half-wave pair.
        float s = c[0] + c[1] + c[2] + c[3] + c[4] + c[5] + c[6] + c[7];
        s += __shfl_xor(s, 16, 32);
        if (tid == 0) bcast = s;                 // == (float)row_count, exact
    }
    __syncthreads();

    // ---- Phase 3: expand mask and stream out ----
    const float total = bcast;
    const float fv    = f[0];
    const float denom = (total > 0.0f) ? (total * fv) : 1.0f;  // safe_denom
    const float inv   = fv / denom;                            // f / (count*f)

#pragma unroll
    for (int k = 0; k < F4_PER_THREAD; ++k) {
        float4 o;
        o.x = ((m >> (4 * k + 0)) & 1u) ? inv : 0.0f;
        o.y = ((m >> (4 * k + 1)) & 1u) ? inv : 0.0f;
        o.z = ((m >> (4 * k + 2)) & 1u) ? inv : 0.0f;
        o.w = ((m >> (4 * k + 3)) & 1u) ? inv : 0.0f;
        yr[k * TPB + tid] = o;
    }
}

extern "C" void kernel_launch(void* const* d_in, const int* in_sizes, int n_in,
                              void* d_out, int out_size, void* d_ws, size_t ws_size,
                              hipStream_t stream)
{
    const float* x = (const float*)d_in[0];   // [16384, 8192] fp32
    const float* f = (const float*)d_in[1];   // [1] fp32 scalar param
    float* y       = (float*)d_out;           // [16384, 8192] fp32

    const int rows = in_sizes[0] / ROW_D;     // 16384
    rownorm_mask_wmma_kernel<<<dim3(rows), dim3(TPB), 0, stream>>>(x, f, y);
}
